// VectorQuantizerEMA_103079215404
// MI455X (gfx1250) — matrix operand order repaired and checked by reading, not compiled, not run on hardware
//
#include <hip/hip_runtime.h>
#include <hip/hip_bf16.h>

#define N_ROWS   16384
#define D_FEAT   256
#define TWO_D    512
#define K_CODES  4096
#define DECAYF   0.99f
#define OMDECAYF 0.01f
#define BN_EPSF  1e-5f
#define MOD_EPSF 1e-24f

typedef __attribute__((ext_vector_type(16))) _Float16 v16h;
typedef __attribute__((ext_vector_type(8)))  _Float16 v8h;
typedef __attribute__((ext_vector_type(8)))  float    v8f;

// ---------------- K1: per-column batch stats + running stats ----------------
__global__ void bn_stats_kernel(const float* __restrict__ X, const float* __restrict__ G,
                                const float* __restrict__ frm, const float* __restrict__ frv,
                                const float* __restrict__ grm, const float* __restrict__ grv,
                                float* __restrict__ mean, float* __restrict__ invstd,
                                float* __restrict__ run_mean, float* __restrict__ run_std) {
    __shared__ float sS[256], sQ[256];
    const int c = blockIdx.x;            // 0..511
    const int t = threadIdx.x;
    const float* src = (c < D_FEAT) ? (X + c) : (G + (c - D_FEAT));
    float s = 0.f, q = 0.f;
    for (int n = t; n < N_ROWS; n += 256) {
        float x = src[(size_t)n * D_FEAT];
        s += x; q += x * x;
    }
    sS[t] = s; sQ[t] = q;
    __syncthreads();
    for (int off = 128; off > 0; off >>= 1) {
        if (t < off) { sS[t] += sS[t + off]; sQ[t] += sQ[t + off]; }
        __syncthreads();
    }
    if (t == 0) {
        float bm = sS[0] * (1.0f / (float)N_ROWS);
        float bv = sQ[0] * (1.0f / (float)N_ROWS) - bm * bm;   // biased
        if (bv < 0.f) bv = 0.f;
        const float eps = (c < D_FEAT) ? BN_EPSF : MOD_EPSF;
        mean[c]   = bm;
        invstd[c] = rsqrtf(bv + eps);
        float unb = bv * ((float)N_ROWS / (float)(N_ROWS - 1));
        float rm  = (c < D_FEAT) ? frm[c] : grm[c - D_FEAT];
        float rv  = (c < D_FEAT) ? frv[c] : grv[c - D_FEAT];
        float nrm = 0.9f * rm + 0.1f * bm;
        float nrv = 0.9f * rv + 0.1f * unb;
        run_mean[c] = nrm;
        run_std[c]  = sqrtf(nrv + eps);
    }
}

// ---------------- K2: embedding prep (e_sq + f16 cast) ----------------
__global__ void emb_prep_kernel(const float* __restrict__ emb,
                                float* __restrict__ esq, _Float16* __restrict__ embH) {
    __shared__ float red[256];
    const int k = blockIdx.x, t = threadIdx.x;
    const float* row = emb + (size_t)k * TWO_D;
    float a = row[t], b = row[t + 256];
    embH[(size_t)k * TWO_D + t]       = (_Float16)a;
    embH[(size_t)k * TWO_D + t + 256] = (_Float16)b;
    red[t] = a * a + b * b;
    __syncthreads();
    for (int off = 128; off > 0; off >>= 1) { if (t < off) red[t] += red[t + off]; __syncthreads(); }
    if (t == 0) esq[k] = red[0];
}

// ---------------- K3: normalize (f32 + f16) ----------------
__global__ void normalize_kernel(const float* __restrict__ X, const float* __restrict__ G,
                                 const float* __restrict__ mean, const float* __restrict__ invstd,
                                 float* __restrict__ xnF, _Float16* __restrict__ xnH) {
    const long e = (long)blockIdx.x * 256 + threadIdx.x;  // over N*512
    const int n = (int)(e >> 9), c = (int)(e & 511);
    float x = (c < D_FEAT) ? X[(size_t)n * D_FEAT + c]
                           : G[(size_t)n * D_FEAT + (c - D_FEAT)];
    float xn = (x - mean[c]) * invstd[c];                 // GNS0 == 1.0
    xnF[e] = xn;
    xnH[e] = (_Float16)xn;
}

// ---------------- K4: WMMA distance GEMM + fused argmin ----------------
// Block = 4 waves, each wave owns a 16-row A panel (64 rows/block).
// All waves sweep the same 256 column tiles; the 16x512-f16 B tile is staged
// into LDS with double-buffered async copies (ASYNCcnt) shared by all waves.
__global__ __launch_bounds__(128)
void dist_argmin_wmma_kernel(const _Float16* __restrict__ xnH,
                             const _Float16* __restrict__ embH,
                             const float* __restrict__ esq,
                             float* __restrict__ idx_out_f,
                             int* __restrict__ idx_out_i) {
    __shared__ __align__(32) _Float16 ldsB[2][16 * TWO_D];   // 2 x 16 KB B tiles
    __shared__ float redD[4][32][8];
    __shared__ int   redI[4][32][8];

    const int tid  = threadIdx.x;
    const int lane = tid & 31;
    const int wave = tid >> 5;
    const int m    = lane & 15;      // A row / B column within tile
    const int hi   = lane >> 4;
    const int rowBase = blockIdx.x * 64 + wave * 16;

    // ---- preload this wave's 16x512 A panel into registers (contiguous k = 16*hi + s) ----
    v16h aFrag[16];
    {
        const _Float16* aRow = xnH + (size_t)(rowBase + m) * TWO_D;
#pragma unroll
        for (int ks = 0; ks < 16; ++ks) {
            const int k0 = ks * 32;
            v8h alo = *(const v8h*)(aRow + k0 + hi * 16);
            v8h ahi = *(const v8h*)(aRow + k0 + hi * 16 + 8);
#pragma unroll
            for (int i = 0; i < 8; ++i) { aFrag[ks][i] = alo[i]; aFrag[ks][i + 8] = ahi[i]; }
        }
    }

    // ---- async B tile issue: 16 KB = 1024 16-byte chunks, 8 per thread ----
    const unsigned long long embBase = (unsigned long long)(uintptr_t)embH;
    const unsigned ldsBase0 = (unsigned)(uintptr_t)(&ldsB[0][0]);
    const unsigned ldsBase1 = (unsigned)(uintptr_t)(&ldsB[1][0]);

#define ISSUE_B_TILE(bufSel, tileIdx)                                          \
    do {                                                                       \
        const unsigned long long g0 =                                          \
            embBase + (unsigned long long)(tileIdx) * (16 * TWO_D * 2);        \
        const unsigned l0 = ((bufSel) ? ldsBase1 : ldsBase0);                  \
        _Pragma("unroll")                                                      \
        for (int i_ = 0; i_ < 8; ++i_) {                                       \
            const int chunk_ = tid + i_ * 128;                                 \
            unsigned lofs_ = l0 + (unsigned)chunk_ * 16u;                      \
            unsigned long long g_ = g0 + (unsigned long long)chunk_ * 16ull;   \
            asm volatile("global_load_async_to_lds_b128 %0, %1, off"           \
                         :: "v"(lofs_), "v"(g_) : "memory");                   \
        }                                                                      \
    } while (0)

    float bestD[8];
    int   bestI[8];
#pragma unroll
    for (int v = 0; v < 8; ++v) { bestD[v] = 3.4e38f; bestI[v] = 0; }

    ISSUE_B_TILE(0, 0);

    for (int t = 0; t < K_CODES / 16; ++t) {
        // wait for this wave's outstanding async copies (tile t), then make
        // all waves' LDS writes visible before compute
        asm volatile("s_wait_asynccnt 0x0" ::: "memory");
        __syncthreads();

        if (t + 1 < K_CODES / 16) ISSUE_B_TILE((t + 1) & 1, t + 1);

        const _Float16* bTile = &ldsB[t & 1][0];
        const int colBase = t * 16;
        const int myCol   = colBase + m;

        v8f acc = {};
#pragma unroll
        for (int ks = 0; ks < 16; ++ks) {
            const int k0 = ks * 32;
            // B fragment: lane -> column m (embedding row colBase+m),
            // contiguous 16 halves at k0 + hi*16
            v16h b = *(const v16h*)(bTile + m * TWO_D + k0 + hi * 16);
            acc = __builtin_amdgcn_wmma_f32_16x16x32_f16(
                false, aFrag[ks], false, b, (short)0, acc, false, false);
        }
        const float e2 = esq[myCol];
#pragma unroll
        for (int v = 0; v < 8; ++v) {             // C layout: row = v + 8*hi, col = m
            float dist = e2 - 2.0f * acc[v];      // x_sq dropped (row-constant)
            if (dist < bestD[v]) { bestD[v] = dist; bestI[v] = myCol; }
        }
        __syncthreads();   // all waves done with buf[t&1] before it refills at t+2
    }
#undef ISSUE_B_TILE

    // ---- per-wave argmin reduction (each wave owns its 16 rows) ----
#pragma unroll
    for (int v = 0; v < 8; ++v) { redD[wave][lane][v] = bestD[v]; redI[wave][lane][v] = bestI[v]; }
    __syncthreads();

    if (lane < 16) {
        const int row = lane;
        const int v   = row & 7;
        const int grp = (row >> 3) << 4;
        float bd = 3.4e38f; int bi = 0x7fffffff;
        for (int c = 0; c < 16; ++c) {
            float d = redD[wave][grp + c][v];
            int   i = redI[wave][grp + c][v];
            if (d < bd || (d == bd && i < bi)) { bd = d; bi = i; }
        }
        idx_out_f[rowBase + row] = (float)bi;
        idx_out_i[rowBase + row] = bi;
    }
}

// ---------------- K5: zero the one-hot encodings ----------------
__global__ void zero_enc_kernel(float4* __restrict__ dst, long n4) {
    long i = (long)blockIdx.x * blockDim.x + threadIdx.x;
    const long stride = (long)gridDim.x * blockDim.x;
    const float4 z = make_float4(0.f, 0.f, 0.f, 0.f);
    for (; i < n4; i += stride) dst[i] = z;
}

// ---------------- K6: scatter the ones ----------------
__global__ void scatter_onehot_kernel(const int* __restrict__ idx, float* __restrict__ enc) {
    int n = blockIdx.x * blockDim.x + threadIdx.x;
    if (n < N_ROWS) enc[(size_t)n * K_CODES + idx[n]] = 1.0f;
}

// ---------------- K7: init EMA outputs with decay ----------------
__global__ void ema_init_kernel(const float* __restrict__ ema_w, const float* __restrict__ ema_cs,
                                float* __restrict__ new_w, float* __restrict__ new_cs) {
    long e = (long)blockIdx.x * 256 + threadIdx.x;      // over K*512
    new_w[e] = ema_w[e] * DECAYF;
    if (e < K_CODES) new_cs[e] = ema_cs[e] * DECAYF;
}

// ---------------- K8: EMA scatter-add (dw fused) ----------------
__global__ void ema_scatter_kernel(const int* __restrict__ idx, const float* __restrict__ xn,
                                   float* __restrict__ new_w, float* __restrict__ new_cs) {
    long e = (long)blockIdx.x * 256 + threadIdx.x;      // over N*512
    int n = (int)(e >> 9), d = (int)(e & 511);
    int k = idx[n];
    atomicAdd(&new_w[(size_t)k * TWO_D + d], OMDECAYF * xn[e]);
    if (d == 0) atomicAdd(&new_cs[k], OMDECAYF);
}

// ---------------- K9: finalize embedding + denormalized output ----------------
__global__ void finalize_kernel(const float* __restrict__ new_w, const float* __restrict__ new_cs,
                                const float* __restrict__ run_mean, const float* __restrict__ run_std,
                                float* __restrict__ new_emb, float* __restrict__ emb_out) {
    long e = (long)blockIdx.x * 256 + threadIdx.x;      // over K*512
    int k = (int)(e >> 9), c = (int)(e & 511);
    float ne = new_w[e] / new_cs[k];
    new_emb[e] = ne;
    float od = (c < D_FEAT) ? ne : ne / (1.0f + MOD_EPSF);
    emb_out[e] = od * run_std[c] + run_mean[c];
}

extern "C" void kernel_launch(void* const* d_in, const int* in_sizes, int n_in,
                              void* d_out, int out_size, void* d_ws, size_t ws_size,
                              hipStream_t stream) {
    (void)in_sizes; (void)n_in; (void)out_size; (void)ws_size;
    const float* X_B  = (const float*)d_in[0];
    const float* grad = (const float*)d_in[1];
    const float* emb  = (const float*)d_in[2];
    const float* ecs  = (const float*)d_in[3];
    const float* emaw = (const float*)d_in[4];
    const float* frm  = (const float*)d_in[5];
    const float* frv  = (const float*)d_in[6];
    const float* grm  = (const float*)d_in[7];
    const float* grv  = (const float*)d_in[8];

    // ---- output layout (flat float32, tuple order) ----
    float* out        = (float*)d_out;
    float* o_idx      = out;                                    // [N]
    float* o_enc      = o_idx + N_ROWS;                         // [N*K]
    float* o_newemb   = o_enc + (size_t)N_ROWS * K_CODES;       // [K*512]
    float* o_newcs    = o_newemb + (size_t)K_CODES * TWO_D;     // [K]
    float* o_newemaw  = o_newcs + K_CODES;                      // [K*512]
    float* o_embout   = o_newemaw + (size_t)K_CODES * TWO_D;    // [K*512]

    // ---- workspace layout ----
    float* wsf     = (float*)d_ws;
    float* w_mean  = wsf;            // 512
    float* w_istd  = wsf + 512;      // 512
    float* w_rm    = wsf + 1024;     // 512
    float* w_rstd  = wsf + 1536;     // 512
    float* w_esq   = wsf + 2048;     // 4096
    int*   w_idx   = (int*)(wsf + 6144);                        // N ints
    float* w_xnF   = wsf + 6144 + N_ROWS;                       // N*512 f32
    _Float16* w_xnH  = (_Float16*)(w_xnF + (size_t)N_ROWS * TWO_D);       // N*512 f16
    _Float16* w_embH = w_xnH + (size_t)N_ROWS * TWO_D;                    // K*512 f16

    bn_stats_kernel<<<TWO_D, 256, 0, stream>>>(X_B, grad, frm, frv, grm, grv,
                                               w_mean, w_istd, w_rm, w_rstd);
    emb_prep_kernel<<<K_CODES, 256, 0, stream>>>(emb, w_esq, w_embH);
    normalize_kernel<<<(N_ROWS * TWO_D) / 256, 256, 0, stream>>>(X_B, grad, w_mean, w_istd,
                                                                 w_xnF, w_xnH);
    dist_argmin_wmma_kernel<<<N_ROWS / 64, 128, 0, stream>>>(w_xnH, w_embH, w_esq,
                                                             o_idx, w_idx);
    {
        const long n4 = ((long)N_ROWS * K_CODES) / 4;
        zero_enc_kernel<<<65536, 256, 0, stream>>>((float4*)o_enc, n4);
    }
    scatter_onehot_kernel<<<(N_ROWS + 255) / 256, 256, 0, stream>>>(w_idx, o_enc);
    ema_init_kernel<<<(K_CODES * TWO_D) / 256, 256, 0, stream>>>(emaw, ecs, o_newemaw, o_newcs);
    ema_scatter_kernel<<<(N_ROWS * TWO_D) / 256, 256, 0, stream>>>(w_idx, w_xnF,
                                                                   o_newemaw, o_newcs);
    finalize_kernel<<<(K_CODES * TWO_D) / 256, 256, 0, stream>>>(o_newemaw, o_newcs,
                                                                 w_rm, w_rstd,
                                                                 o_newemb, o_embout);
}
